// CategoricalEncoder_60576218743446
// MI455X (gfx1250) — matrix operand order repaired and checked by reading, not compiled, not run on hardware
//
#include <hip/hip_runtime.h>
#include <hip/hip_fp16.h>

typedef __attribute__((ext_vector_type(16))) _Float16 v16h;
typedef __attribute__((ext_vector_type(8)))  float    v8f;
typedef __attribute__((ext_vector_type(4)))  float    v4f;

#define N_STATES 32
#define STOCH    32
#define EMBED    128

// --- deterministic stateless RNG (PCG hash) -> Gumbel noise -----------------
__device__ __forceinline__ unsigned pcg_hash(unsigned v) {
    v = v * 747796405u + 2891336453u;
    unsigned w = ((v >> ((v >> 28u) + 4u)) ^ v) * 277803737u;
    return (w >> 22u) ^ w;
}

// ---------------------------------------------------------------------------
// Kernel 1: repack fp32 codebook [N, S=32, E=128] into f16 tiles stored in the
// per-lane register order this file uses for the WMMA B operand (32x16, f16), contiguous K:
//   lanes 0-15 : K = 0..15  (half-slot j -> K = j),  col = lane & 15
//   lanes 16-31: K = 16..31 (half-slot j -> K = 16+j)
// pre[((n*8 + etile)*32 + lane)*16 + j]
// ---------------------------------------------------------------------------
__global__ void pack_codebook_kernel(const float* __restrict__ cb,
                                     _Float16* __restrict__ pre) {
    int t    = blockIdx.x * blockDim.x + threadIdx.x;   // 32*8*32 = 8192 threads
    int lane = t & 31;
    int et   = (t >> 5) & 7;
    int n    = t >> 8;
    int col  = lane & 15;
    int kb   = (lane < 16) ? 0 : 16;
    const float* src = cb + (size_t)n * (STOCH * EMBED) + et * 16 + col;
    v16h b;
#pragma unroll
    for (int j = 0; j < 16; ++j)
        b[j] = (_Float16)src[(size_t)(kb + j) * EMBED];
    *(v16h*)(pre + (size_t)t * 16) = b;                 // 2x global_store_b128
}

// ---------------------------------------------------------------------------
// Kernel 2: one wave per (16-row tile of BT, n). Gumbel-max sample -> one-hot
// A (16x32 f16) -> 8x V_WMMA_F32_16X16X32_F16 against packed codebook tiles.
// ---------------------------------------------------------------------------
__global__ void __launch_bounds__(256)
cat_encode_kernel(const float* __restrict__ x,
                  const _Float16* __restrict__ pre,
                  float* __restrict__ out) {
    const int lane    = threadIdx.x & 31;
    const int wave    = threadIdx.x >> 5;
    const int W       = blockIdx.x * 8 + wave;          // global wave id
    const int n       = W & (N_STATES - 1);
    const int rowbase = (W >> 5) * 16;                  // 16 (b,t) rows per wave
    const int r       = lane & 15;                      // row within tile
    const int hv      = lane >> 4;                      // which 16 classes

    // ---- load 16 contiguous logits for (row, n, half) : 4x global_load_b128
    const float* xp = x + (size_t)(rowbase + r) * (N_STATES * STOCH)
                        + n * STOCH + hv * 16;          // 64B aligned
    v4f q0 = *(const v4f*)(xp + 0);
    v4f q1 = *(const v4f*)(xp + 4);
    v4f q2 = *(const v4f*)(xp + 8);
    v4f q3 = *(const v4f*)(xp + 12);
    float lg[16] = { q0[0], q0[1], q0[2], q0[3],
                     q1[0], q1[1], q1[2], q1[3],
                     q2[0], q2[1], q2[2], q2[3],
                     q3[0], q3[1], q3[2], q3[3] };

    // ---- Gumbel-max over local 16 classes
    unsigned seedbase = (((unsigned)(rowbase + r) * 32u + (unsigned)n) * 32u)
                        + (unsigned)(hv * 16);
    float best = -3.4e38f;
    int   bidx = hv * 16;
#pragma unroll
    for (int k = 0; k < 16; ++k) {
        unsigned h = pcg_hash(seedbase + (unsigned)k);
        float u = ((float)(h >> 8) + 0.5f) * (1.0f / 16777216.0f); // (0,1)
        float g = -__logf(-__logf(u));
        float v = lg[k] + g;
        if (v > best) { best = v; bidx = hv * 16 + k; }
    }
    // ---- merge with partner lane (same row, other 16 classes)
    float obest = __shfl_xor(best, 16);
    int   oidx  = __shfl_xor(bidx, 16);
    if (obest > best || (obest == best && oidx < bidx)) { best = obest; bidx = oidx; }

    // ---- build one-hot A (16x32) with the same contiguous K order as packed B
    // lanes 0-15 : slot i -> K = i
    // lanes 16-31: slot i -> K = 16+i
    v16h a;
#pragma unroll
    for (int i = 0; i < 16; ++i) {
        int K = (lane < 16) ? i
                            : i + 16;
        a[i] = (K == bidx) ? (_Float16)1.0f : (_Float16)0.0f;
    }

    // ---- 8 E-tiles of 16: one WMMA each, store D (16x16 f32)
    float*          op = out + (size_t)rowbase * (N_STATES * EMBED) + n * EMBED;
    const _Float16* bp = pre + ((size_t)(n * 8) * 32 + lane) * 16;
#pragma unroll
    for (int et = 0; et < 8; ++et) {
        v16h bm = *(const v16h*)(bp + (size_t)et * 32 * 16); // 2x b128, coalesced
        v8f c = {};
        c = __builtin_amdgcn_wmma_f32_16x16x32_f16(
                /*neg_a=*/false, a, /*neg_b=*/false, bm,
                /*c_mod=*/(short)0, c, /*reuse_a=*/false, /*reuse_b=*/false);
        // D layout: VGPR v -> row v (lanes 0-15) / row v+8 (lanes 16-31), col = lane&15
#pragma unroll
        for (int v = 0; v < 8; ++v) {
            int ro = (lane < 16) ? v : v + 8;
            op[(size_t)ro * (N_STATES * EMBED) + et * 16 + r] = c[v];
        }
    }
}

extern "C" void kernel_launch(void* const* d_in, const int* in_sizes, int n_in,
                              void* d_out, int out_size, void* d_ws, size_t ws_size,
                              hipStream_t stream) {
    const float* x   = (const float*)d_in[0];   // [B,T,N*S] fp32
    const float* cb  = (const float*)d_in[1];   // [N,S,E]   fp32
    float*       out = (float*)d_out;           // [B,T,N*E] fp32
    _Float16*    pre = (_Float16*)d_ws;         // 256 KB packed f16 codebook

    int BT = in_sizes[0] / (N_STATES * STOCH);  // 8192 for B=32,T=256

    // repack codebook into WMMA-B lane order (8192 threads)
    pack_codebook_kernel<<<32, 256, 0, stream>>>(cb, pre);

    // one wave per (16-row tile, n): (BT/16)*32 waves, 8 waves per block
    int nblocks = (BT / 16) * N_STATES / 8;
    cat_encode_kernel<<<nblocks, 256, 0, stream>>>(x, pre, out);
}
